// LSTM_3143916060861
// MI455X (gfx1250) — matrix operand order repaired and checked by reading, not compiled, not run on hardware
//
#include <hip/hip_runtime.h>
#include <math.h>

// ---------------------------------------------------------------------------
// 2-layer LSTM (H=1024, B=64, T=512) for gfx1250.
//  * batch split across 4 workgroups (16 rows each) -> zero inter-WG sync
//  * per-wave: 2 hidden-column tiles; i/f/g/o gate tiles computed by the same
//    wave so the cell state c1/c2 stays in WMMA-accumulator VGPRs forever
//  * h1/h2 state in LDS as bf16 (64 KB), weights converted to bf16 in d_ws
//  * all matmuls via v_wmma_f32_16x16x32_bf16
// ---------------------------------------------------------------------------

typedef __attribute__((ext_vector_type(16))) __bf16 bf16x16;
typedef __attribute__((ext_vector_type(8)))  __bf16 bf16x8;
typedef __attribute__((ext_vector_type(8)))  float  f32x8;

#define HID    1024
#define BATCH  64
#define MB     16        // batch rows per workgroup
#define NWAVES 32        // waves per workgroup (1024 threads)
#define KTILE  32        // K per wmma
#define NKT    (HID / KTILE)

union ABfrag { bf16x16 v; bf16x8 h[2]; };

__device__ __forceinline__ float sig_(float x) { return 1.0f / (1.0f + __expf(-x)); }

// One 16x16xK (K=1024) accumulation of 4 gate tiles sharing the same A rows.
// ha : per-lane A base (LDS, bf16), already offset by rowA*HID + kA
// wg0..wg3 : per-lane B base (global bf16 weights), already offset by
//            (gate*HID + j0 + col)*HID + kB
__device__ __forceinline__ void gemm4(const __bf16* __restrict__ ha,
                                      const __bf16* __restrict__ wg0,
                                      const __bf16* __restrict__ wg1,
                                      const __bf16* __restrict__ wg2,
                                      const __bf16* __restrict__ wg3,
                                      f32x8 acc[4]) {
#pragma unroll 4
  for (int kt = 0; kt < NKT; ++kt) {
    const int k0 = kt * KTILE;
    ABfrag A, B0, B1, B2, B3;
    // A: lanes 0-15 hold K {0..15}, lanes 16-31 hold K {16..31}
    A.h[0] = *(const bf16x8*)(ha + k0);
    A.h[1] = *(const bf16x8*)(ha + k0 + 8);
    // B: each lane holds 16 contiguous K values of its N column
    B0.h[0] = *(const bf16x8*)(wg0 + k0); B0.h[1] = *(const bf16x8*)(wg0 + k0 + 8);
    B1.h[0] = *(const bf16x8*)(wg1 + k0); B1.h[1] = *(const bf16x8*)(wg1 + k0 + 8);
    B2.h[0] = *(const bf16x8*)(wg2 + k0); B2.h[1] = *(const bf16x8*)(wg2 + k0 + 8);
    B3.h[0] = *(const bf16x8*)(wg3 + k0); B3.h[1] = *(const bf16x8*)(wg3 + k0 + 8);
    acc[0] = __builtin_amdgcn_wmma_f32_16x16x32_bf16(false, A.v, false, B0.v, (short)0, acc[0], false, false);
    acc[1] = __builtin_amdgcn_wmma_f32_16x16x32_bf16(false, A.v, false, B1.v, (short)0, acc[1], false, false);
    acc[2] = __builtin_amdgcn_wmma_f32_16x16x32_bf16(false, A.v, false, B2.v, (short)0, acc[2], false, false);
    acc[3] = __builtin_amdgcn_wmma_f32_16x16x32_bf16(false, A.v, false, B3.v, (short)0, acc[3], false, false);
  }
}

__global__ void cvt_bf16_kernel(const float* __restrict__ src,
                                __bf16* __restrict__ dst, int n) {
  int i = blockIdx.x * blockDim.x + threadIdx.x;
  const int stride = gridDim.x * blockDim.x;
  for (; i < n; i += stride) dst[i] = (__bf16)src[i];
}

__global__ void
__launch_bounds__(NWAVES * 32)
lstm_kernel(const float* __restrict__ x,        // [T*BATCH]
            const float* __restrict__ Wih1,     // [4*HID]  (input dim == 1)
            const float* __restrict__ bih1, const float* __restrict__ bhh1,
            const float* __restrict__ bih2, const float* __restrict__ bhh2,
            const float* __restrict__ Wlin,     // [HID]
            const float* __restrict__ blin,     // [1]
            const __bf16* __restrict__ Whh1,    // [4*HID, HID] bf16 row-major
            const __bf16* __restrict__ Wih2,
            const __bf16* __restrict__ Whh2,
            float* __restrict__ out,            // [T*BATCH]
            int T) {
  extern __shared__ __align__(16) char smem[];
  __bf16* h1s = (__bf16*)smem;                       // [MB][HID] bf16
  __bf16* h2s = (__bf16*)(smem + MB * HID * 2);      // [MB][HID] bf16

  const int tid    = threadIdx.x;
  const int lane   = tid & 31;
  const int wave   = tid >> 5;
  const int batch0 = blockIdx.x * MB;

  const int col    = lane & 15;   // N-within-tile (B/C/D frag column)
  const int hiHalf = lane >> 4;   // lane group 0/1
  const int kA     = hiHalf * 16; // A-frag K sub-offset
  const int kB     = hiHalf * 16; // B-frag K sub-offset
  const int rowA   = col;         // A-frag M row

  // zero recurrent state in LDS
  for (int i = tid; i < 2 * MB * HID; i += blockDim.x) ((__bf16*)smem)[i] = (__bf16)0.0f;
  __syncthreads();

  // cell state lives in accumulator-layout VGPRs for the whole sequence
  f32x8 c1f[2], c2f[2];
#pragma unroll
  for (int jj = 0; jj < 2; ++jj)
#pragma unroll
    for (int r = 0; r < 8; ++r) { c1f[jj][r] = 0.0f; c2f[jj][r] = 0.0f; }

  const float bl = blin[0];

  for (int t = 0; t < T; ++t) {
    // per-lane copy of this step's inputs for the 8 batch rows it accumulates
    float xr[8];
#pragma unroll
    for (int r = 0; r < 8; ++r) xr[r] = x[t * BATCH + batch0 + r + hiHalf * 8];

    // ================= layer 1: gates1 = b1 + x*Wih1 + h1 @ Whh1^T =========
    f32x8 h1v[2];
#pragma unroll
    for (int jj = 0; jj < 2; ++jj) {
      const int j0 = (wave * 2 + jj) * 16;
      f32x8 acc[4];
#pragma unroll
      for (int g = 0; g < 4; ++g) {
        const int n = g * HID + j0 + col;
        const float bsum = bih1[n] + bhh1[n];
        const float w    = Wih1[n];
#pragma unroll
        for (int r = 0; r < 8; ++r) acc[g][r] = bsum + xr[r] * w;
      }
      gemm4(h1s + rowA * HID + kA,
            Whh1 + (size_t)(0 * HID + j0 + col) * HID + kB,
            Whh1 + (size_t)(1 * HID + j0 + col) * HID + kB,
            Whh1 + (size_t)(2 * HID + j0 + col) * HID + kB,
            Whh1 + (size_t)(3 * HID + j0 + col) * HID + kB, acc);
#pragma unroll
      for (int r = 0; r < 8; ++r) {
        const float ig = sig_(acc[0][r]);
        const float fg = sig_(acc[1][r]);
        const float gg = tanhf(acc[2][r]);
        const float og = sig_(acc[3][r]);
        const float c  = fg * c1f[jj][r] + ig * gg;
        c1f[jj][r] = c;
        h1v[jj][r] = og * tanhf(c);
      }
    }
    __syncthreads();  // all reads of old h1 complete
#pragma unroll
    for (int jj = 0; jj < 2; ++jj) {
      const int j0 = (wave * 2 + jj) * 16;
#pragma unroll
      for (int r = 0; r < 8; ++r)
        h1s[(r + hiHalf * 8) * HID + j0 + col] = (__bf16)h1v[jj][r];
    }
    __syncthreads();  // new h1 visible

    // ========= layer 2: gates2 = b2 + h1 @ Wih2^T + h2 @ Whh2^T ===========
    f32x8 h2v[2];
#pragma unroll
    for (int jj = 0; jj < 2; ++jj) {
      const int j0 = (wave * 2 + jj) * 16;
      f32x8 acc[4];
#pragma unroll
      for (int g = 0; g < 4; ++g) {
        const int n = g * HID + j0 + col;
        const float bsum = bih2[n] + bhh2[n];
#pragma unroll
        for (int r = 0; r < 8; ++r) acc[g][r] = bsum;
      }
      gemm4(h1s + rowA * HID + kA,
            Wih2 + (size_t)(0 * HID + j0 + col) * HID + kB,
            Wih2 + (size_t)(1 * HID + j0 + col) * HID + kB,
            Wih2 + (size_t)(2 * HID + j0 + col) * HID + kB,
            Wih2 + (size_t)(3 * HID + j0 + col) * HID + kB, acc);
      gemm4(h2s + rowA * HID + kA,
            Whh2 + (size_t)(0 * HID + j0 + col) * HID + kB,
            Whh2 + (size_t)(1 * HID + j0 + col) * HID + kB,
            Whh2 + (size_t)(2 * HID + j0 + col) * HID + kB,
            Whh2 + (size_t)(3 * HID + j0 + col) * HID + kB, acc);
#pragma unroll
      for (int r = 0; r < 8; ++r) {
        const float ig = sig_(acc[0][r]);
        const float fg = sig_(acc[1][r]);
        const float gg = tanhf(acc[2][r]);
        const float og = sig_(acc[3][r]);
        const float c  = fg * c2f[jj][r] + ig * gg;
        c2f[jj][r] = c;
        h2v[jj][r] = og * tanhf(c);
      }
    }
    __syncthreads();  // all reads of old h2 complete
#pragma unroll
    for (int jj = 0; jj < 2; ++jj) {
      const int j0 = (wave * 2 + jj) * 16;
#pragma unroll
      for (int r = 0; r < 8; ++r)
        h2s[(r + hiHalf * 8) * HID + j0 + col] = (__bf16)h2v[jj][r];
    }
    __syncthreads();  // new h2 visible

    // ================= out[t,b] = h2[b,:] . Wlin + blin ====================
    if (wave < MB) {
      const int b = wave;
      float sum = 0.0f;
#pragma unroll 4
      for (int kk = 0; kk < HID / 32; ++kk) {
        const int k = kk * 32 + lane;
        sum += (float)h2s[b * HID + k] * Wlin[k];
      }
#pragma unroll
      for (int off = 16; off > 0; off >>= 1) sum += __shfl_xor(sum, off, 32);
      if (lane == 0) out[t * BATCH + batch0 + b] = sum + bl;
    }
    // next step's writers cannot pass their first __syncthreads until the
    // output waves arrive, so the h2 read above is race-free
  }
}

extern "C" void kernel_launch(void* const* d_in, const int* in_sizes, int n_in,
                              void* d_out, int out_size, void* d_ws, size_t ws_size,
                              hipStream_t stream) {
  const float* x    = (const float*)d_in[0];
  const float* Wih1 = (const float*)d_in[1];
  const float* Whh1 = (const float*)d_in[2];
  const float* bih1 = (const float*)d_in[3];
  const float* bhh1 = (const float*)d_in[4];
  const float* Wih2 = (const float*)d_in[5];
  const float* Whh2 = (const float*)d_in[6];
  const float* bih2 = (const float*)d_in[7];
  const float* bhh2 = (const float*)d_in[8];
  const float* Wlin = (const float*)d_in[9];
  const float* blin = (const float*)d_in[10];

  const int T = in_sizes[0] / BATCH;  // 512

  // bf16 weight copies in workspace (3 x 8 MB)
  const int WN = 4 * HID * HID;
  __bf16* Whh1b = (__bf16*)d_ws;
  __bf16* Wih2b = Whh1b + WN;
  __bf16* Whh2b = Wih2b + WN;
  cvt_bf16_kernel<<<1024, 256, 0, stream>>>(Whh1, Whh1b, WN);
  cvt_bf16_kernel<<<1024, 256, 0, stream>>>(Wih2, Wih2b, WN);
  cvt_bf16_kernel<<<1024, 256, 0, stream>>>(Whh2, Whh2b, WN);

  const size_t shmem = (size_t)2 * MB * HID * sizeof(__bf16);  // 64 KB
  lstm_kernel<<<BATCH / MB, NWAVES * 32, shmem, stream>>>(
      x, Wih1, bih1, bhh1, bih2, bhh2, Wlin, blin,
      Whh1b, Wih2b, Whh2b, (float*)d_out, T);
}
